// EGCFModel_51762945851845
// MI455X (gfx1250) — matrix operand order repaired and checked by reading, not compiled, not run on hardware
//
#include <hip/hip_runtime.h>
#include <math.h>

// EGCF model for MI455X (gfx1250, wave32).
// Dense GEMMs via v_wmma_f32_16x16x32_f16 (f32->f16 convert in registers,
// f32 accumulate). Sparse GCN aggregation via global_atomic_add_f32 scatter.
// d_out doubles as the "ne = concat(nn, ee)" buffer, which is also exactly
// the required output layout gu|gi|ee.

#define DD       128
#define NUSERS   40000
#define NITEMS   20000
#define NNODE    60000       // N
#define NEDGE    200000      // NE (edge-as-node count)
#define NTOT     260000      // N + NE
#define EG       800000      // edges per graph

typedef __attribute__((ext_vector_type(16))) _Float16 v16h;
typedef __attribute__((ext_vector_type(8)))  float    v8f;

__global__ __launch_bounds__(256) void fill_f32(float* __restrict__ p, float v, int n) {
  int i = blockIdx.x * 256 + threadIdx.x;
  if (i < n) p[i] = v;
}

__global__ __launch_bounds__(256) void count_deg(const int* __restrict__ col,
                                                 float* __restrict__ deg, int e) {
  int i = blockIdx.x * 256 + threadIdx.x;
  if (i < e) atomicAdd(&deg[col[i]], 1.0f);
}

__global__ __launch_bounds__(256) void rsqrt_inplace(float* __restrict__ p, int n) {
  int i = blockIdx.x * 256 + threadIdx.x;
  if (i < n) p[i] = rsqrtf(p[i]);
}

// Out[M x 128] = act( [A0 | A1][M x (KT*32)] @ W^T + bias ),  W is [128 x K] row-major.
// Logical A columns 0..127 come from A0, 128..255 from A1 (attention concat).
// act: 0 = none, 1 = tanh, 2 = relu.
// Block = 256 threads = 8 waves; wave w owns output columns [16w, 16w+16).
// B fragments (one per K-tile) are loaded once and reused across a grid-stride
// loop over 16-row M tiles. Fragment layouts per CDNA5 ISA 7.12.2.
template <int KT>
__global__ __launch_bounds__(256)
void gemm_wmma(const float* __restrict__ A0, const float* __restrict__ A1,
               const float* __restrict__ W, const float* __restrict__ bias,
               float* __restrict__ Out, int M, int act) {
  const int K    = KT * 32;
  const int lane = threadIdx.x & 31;
  const int wave = threadIdx.x >> 5;
  const int nb   = wave * 16;      // column base for this wave
  const int hi   = lane >> 4;      // K-subgroup select for A
  const int l15  = lane & 15;

  // B: lane -> N = nb + l15, element j -> K = kt*32 + 8*hi + j (j<8), +16 for j>=8.  B[k][n] = W[n][k].
  v16h B[KT];
#pragma unroll
  for (int kt = 0; kt < KT; ++kt) {
    int k = kt * 32 + 8 * hi;
#pragma unroll
    for (int j = 0; j < 16; ++j)
      B[kt][j] = (_Float16)W[(size_t)(nb + l15) * K + k + (j < 8 ? j : j + 8)];
  }
  const float bv = bias ? bias[nb + l15] : 0.0f;

  const int ntiles = M >> 4;
  for (int t = blockIdx.x; t < ntiles; t += gridDim.x) {
    const int m = t * 16 + l15;                     // A row for this lane
    const float* r0 = A0 + (size_t)m * DD;
    const float* r1 = A1 ? (A1 + (size_t)m * DD) : nullptr;

    v8f acc = {0.f, 0.f, 0.f, 0.f, 0.f, 0.f, 0.f, 0.f};
#pragma unroll
    for (int kt = 0; kt < KT; ++kt) {
      // A elements 0..7 -> K = kt*32 + 8*hi + j ; 8..15 -> +16.
      const int c0 = kt * 32 + 8 * hi;
      const int c1 = c0 + 16;
      const float* p0 = (c0 < DD) ? (r0 + c0) : (r1 + (c0 - DD));
      const float* p1 = (c1 < DD) ? (r0 + c1) : (r1 + (c1 - DD));
      v16h a;
#pragma unroll
      for (int j = 0; j < 8; ++j) {
        a[j]     = (_Float16)p0[j];
        a[8 + j] = (_Float16)p1[j];
      }
      acc = __builtin_amdgcn_wmma_f32_16x16x32_f16(false, a, false, B[kt],
                                                   (short)0, acc, false, false);
    }
    // D: lane -> column n = nb + l15, VGPR r -> row t*16 + r + 8*hi.
    const int n  = nb + l15;
    const int mr = t * 16 + 8 * hi;
#pragma unroll
    for (int r = 0; r < 8; ++r) {
      float v = acc[r] + bv;
      if (act == 1)      v = tanhf(v);
      else if (act == 2) v = fmaxf(v, 0.0f);
      Out[(size_t)(mr + r) * DD + n] = v;
    }
  }
}

// out[i,:] = h[i,:] * dinv[i]^2 + bias   (self-loop message + bias)
__global__ __launch_bounds__(256)
void conv_init(const float* __restrict__ h, const float* __restrict__ dinv,
               const float* __restrict__ bias, float* __restrict__ out, int M) {
  size_t i = (size_t)blockIdx.x * 256 + threadIdx.x;
  if (i >= (size_t)M * DD) return;
  int row = (int)(i >> 7), c = (int)(i & 127);
  float di = dinv[row];
  out[i] = h[i] * di * di + bias[c];
}

// out[col[e],:] += h[row[e],:] * dinv[row]*dinv[col]   (one wave per edge)
__global__ __launch_bounds__(256)
void scatter_edges(const float* __restrict__ h, const int* __restrict__ rows,
                   const int* __restrict__ cols, const float* __restrict__ dinv,
                   float* __restrict__ out, int e) {
  int edge = blockIdx.x * 8 + (threadIdx.x >> 5);
  int lane = threadIdx.x & 31;
  if (edge >= e) return;
  int r = rows[edge], c = cols[edge];
  float w = dinv[r] * dinv[c];
  float4 v = ((const float4*)(h + (size_t)r * DD))[lane];
  float* op = out + (size_t)c * DD + lane * 4;
  atomicAdd(op + 0, v.x * w);
  atomicAdd(op + 1, v.y * w);
  atomicAdd(op + 2, v.z * w);
  atomicAdd(op + 3, v.w * w);
}

// logits = h @ W2^T + b2 ; p = softmax(logits / 0.5) ; out = p0*a + p1*c.
// One wave per row; 4 columns per lane; cross-lane shfl_xor reduction.
__global__ __launch_bounds__(256)
void att_mix(const float* __restrict__ a, const float* __restrict__ c,
             const float* __restrict__ h, const float* __restrict__ W2,
             const float* __restrict__ b2, float* __restrict__ out, int M) {
  int row  = blockIdx.x * 8 + (threadIdx.x >> 5);
  int lane = threadIdx.x & 31;
  if (row >= M) return;
  float4 hv = ((const float4*)(h + (size_t)row * DD))[lane];
  float4 w0 = ((const float4*)(W2))[lane];
  float4 w1 = ((const float4*)(W2 + DD))[lane];
  float l0 = hv.x * w0.x + hv.y * w0.y + hv.z * w0.z + hv.w * w0.w;
  float l1 = hv.x * w1.x + hv.y * w1.y + hv.z * w1.z + hv.w * w1.w;
#pragma unroll
  for (int off = 16; off > 0; off >>= 1) {
    l0 += __shfl_xor(l0, off, 32);
    l1 += __shfl_xor(l1, off, 32);
  }
  l0 = (l0 + b2[0]) * 2.0f;   // 1/TEMP = 2
  l1 = (l1 + b2[1]) * 2.0f;
  float mx = fmaxf(l0, l1);
  float e0 = __expf(l0 - mx), e1 = __expf(l1 - mx);
  float p0 = e0 / (e0 + e1), p1 = 1.0f - p0;
  float4 av = ((const float4*)(a + (size_t)row * DD))[lane];
  float4 cv = ((const float4*)(c + (size_t)row * DD))[lane];
  float4 o;
  o.x = p0 * av.x + p1 * cv.x;
  o.y = p0 * av.y + p1 * cv.y;
  o.z = p0 * av.z + p1 * cv.z;
  o.w = p0 * av.w + p1 * cv.w;
  ((float4*)(out + (size_t)row * DD))[lane] = o;
}

static inline unsigned cdivu(size_t a, size_t b) { return (unsigned)((a + b - 1) / b); }

extern "C" void kernel_launch(void* const* d_in, const int* in_sizes, int n_in,
                              void* d_out, int out_size, void* d_ws, size_t ws_size,
                              hipStream_t stream) {
  (void)in_sizes; (void)n_in; (void)out_size; (void)ws_size;
  const float* Gu    = (const float*)d_in[0];
  const float* Gi    = (const float*)d_in[1];
  const float* Ge    = (const float*)d_in[2];
  const float* Wp    = (const float*)d_in[3];
  const float* bp    = (const float*)d_in[4];
  const float* W_nn  = (const float*)d_in[5];
  const float* b_nn  = (const float*)d_in[6];
  const float* W_ee  = (const float*)d_in[7];
  const float* b_ee  = (const float*)d_in[8];
  const float* W_ne  = (const float*)d_in[9];
  const float* b_ne  = (const float*)d_in[10];
  const float* Wa1_n = (const float*)d_in[11];
  const float* ba1_n = (const float*)d_in[12];
  const float* Wa2_n = (const float*)d_in[13];
  const float* ba2_n = (const float*)d_in[14];
  const float* Wa1_e = (const float*)d_in[15];
  const float* ba1_e = (const float*)d_in[16];
  const float* Wa2_e = (const float*)d_in[17];
  const float* ba2_e = (const float*)d_in[18];
  const int*   ei    = (const int*)d_in[19];   // [2, E] node-node
  const int*   eei   = (const int*)d_in[20];   // [2, E] edge-edge
  const int*   nei   = (const int*)d_in[21];   // [2, E] node+edge

  const size_t SZ_N = (size_t)NNODE * DD;
  const size_t SZ_E = (size_t)NEDGE * DD;
  const size_t SZ_T = (size_t)NTOT * DD;

  // d_out = [nn (N x 128) | ee (NE x 128)] = ne buffer = final gu|gi|ee output.
  float* out = (float*)d_out;
  float* nn  = out;
  float* ee  = out + SZ_N;

  float* ws     = (float*)d_ws;
  float* nnc    = ws;                 // conv(nn)           N x 128
  float* eec    = nnc + SZ_N;         // conv(ee)           NE x 128
  float* nec    = eec + SZ_E;         // conv(ne)           (N+NE) x 128
  float* hbuf   = nec + SZ_T;         // GEMM / attention-h scratch, (N+NE) x 128
  float* dinv_n = hbuf + SZ_T;        // N
  float* dinv_e = dinv_n + NNODE;     // NE
  float* dinv_t = dinv_e + NEDGE;     // N+NE

  const dim3 blk(256);
  const unsigned gblk = 2048;         // grid-stride GEMM blocks

  // ---- degrees (constant across layers): deg = 1 (self loop) + count(col) ----
  fill_f32<<<cdivu(NNODE, 256), blk, 0, stream>>>(dinv_n, 1.0f, NNODE);
  fill_f32<<<cdivu(NEDGE, 256), blk, 0, stream>>>(dinv_e, 1.0f, NEDGE);
  fill_f32<<<cdivu(NTOT, 256), blk, 0, stream>>>(dinv_t, 1.0f, NTOT);
  count_deg<<<cdivu(EG, 256), blk, 0, stream>>>(ei  + EG, dinv_n, EG);
  count_deg<<<cdivu(EG, 256), blk, 0, stream>>>(eei + EG, dinv_e, EG);
  count_deg<<<cdivu(EG, 256), blk, 0, stream>>>(nei + EG, dinv_t, EG);
  rsqrt_inplace<<<cdivu(NNODE, 256), blk, 0, stream>>>(dinv_n, NNODE);
  rsqrt_inplace<<<cdivu(NEDGE, 256), blk, 0, stream>>>(dinv_e, NEDGE);
  rsqrt_inplace<<<cdivu(NTOT, 256), blk, 0, stream>>>(dinv_t, NTOT);

  // ---- init: nn = concat(Gu, Gi); ee = tanh(Ge @ Wp^T + bp) ----
  hipMemcpyAsync(nn, Gu, (size_t)NUSERS * DD * sizeof(float), hipMemcpyDeviceToDevice, stream);
  hipMemcpyAsync(nn + (size_t)NUSERS * DD, Gi, (size_t)NITEMS * DD * sizeof(float),
                 hipMemcpyDeviceToDevice, stream);
  gemm_wmma<4><<<gblk, blk, 0, stream>>>(Ge, nullptr, Wp, bp, ee, NEDGE, /*tanh*/1);

  for (int l = 0; l < 2; ++l) {
    const float* Wnl = W_nn + (size_t)l * DD * DD;
    const float* Wel = W_ee + (size_t)l * DD * DD;
    const float* Wtl = W_ne + (size_t)l * DD * DD;

    // nnc = gcn_conv(nn)
    gemm_wmma<4><<<gblk, blk, 0, stream>>>(nn, nullptr, Wnl, nullptr, hbuf, NNODE, 0);
    conv_init<<<cdivu(SZ_N, 256), blk, 0, stream>>>(hbuf, dinv_n, b_nn + l * DD, nnc, NNODE);
    scatter_edges<<<cdivu(EG, 8), blk, 0, stream>>>(hbuf, ei, ei + EG, dinv_n, nnc, EG);

    // eec = gcn_conv(ee)
    gemm_wmma<4><<<gblk, blk, 0, stream>>>(ee, nullptr, Wel, nullptr, hbuf, NEDGE, 0);
    conv_init<<<cdivu(SZ_E, 256), blk, 0, stream>>>(hbuf, dinv_e, b_ee + l * DD, eec, NEDGE);
    scatter_edges<<<cdivu(EG, 8), blk, 0, stream>>>(hbuf, eei, eei + EG, dinv_e, eec, EG);

    // nec = gcn_conv(ne)   (ne lives in d_out = [nn|ee])
    gemm_wmma<4><<<gblk, blk, 0, stream>>>(out, nullptr, Wtl, nullptr, hbuf, NTOT, 0);
    conv_init<<<cdivu(SZ_T, 256), blk, 0, stream>>>(hbuf, dinv_t, b_ne + l * DD, nec, NTOT);
    scatter_edges<<<cdivu(EG, 8), blk, 0, stream>>>(hbuf, nei, nei + EG, dinv_t, nec, EG);

    // nn = att_fuse(nnc, nec[:N])   -> writes d_out[:N]
    gemm_wmma<8><<<gblk, blk, 0, stream>>>(nnc, nec, Wa1_n, ba1_n, hbuf, NNODE, /*relu*/2);
    att_mix<<<cdivu(NNODE, 8), blk, 0, stream>>>(nnc, nec, hbuf, Wa2_n, ba2_n, nn, NNODE);

    // ee = att_fuse(eec, nec[N:])   -> writes d_out[N:]
    gemm_wmma<8><<<gblk, blk, 0, stream>>>(eec, nec + SZ_N, Wa1_e, ba1_e, hbuf, NEDGE, /*relu*/2);
    att_mix<<<cdivu(NEDGE, 8), blk, 0, stream>>>(eec, nec + SZ_N, hbuf, Wa2_e, ba2_e, ee, NEDGE);
  }
  // d_out now holds [gu | gi | ee] — exactly the reference output.
}
